// GlobalSparseAttn_48163763258047
// MI455X (gfx1250) — matrix operand order repaired and checked by reading, not compiled, not run on hardware
//
#include <hip/hip_runtime.h>
#include <hip/hip_bf16.h>
#include <hip/hip_fp16.h>

typedef __attribute__((ext_vector_type(16))) _Float16 v16h;
typedef __attribute__((ext_vector_type(8)))  _Float16 v8h;
typedef __attribute__((ext_vector_type(4)))  _Float16 v4h;
typedef __attribute__((ext_vector_type(8)))  float    v8f;

#define DIM   384
#define HEADS 8
#define HD    48
#define HDP   64      // hd padded to 64 for K=32 WMMA steps
#define BATCH 16
#define HH    56
#define NTOK  3136    // 56*56
#define NSUB  784     // 28*28
#define NPAD  800     // 784 padded to 25*32 key blocks
#define QTILES 49     // 784/16
#define BK    32
#define ATT_SCALE 0.14433756729740643f  // 48^-0.5
#define LN_EPS 1e-5f

__device__ __forceinline__ v16h frag2(const _Float16* p0, const _Float16* p1) {
  union { v16h v; v8h h[2]; } u;
  u.h[0] = *(const v8h*)p0;
  u.h[1] = *(const v8h*)p1;
  return u.v;
}
__device__ __forceinline__ v8f wmma16(v16h a, v16h b, v8f c) {
  // D = A(16x32 f16) * B(32x16 f16) + C(16x16 f32)
  return __builtin_amdgcn_wmma_f32_16x16x32_f16(false, a, false, b, (short)0, c,
                                                false, false);
}
__device__ __forceinline__ v8f zero8() {
  v8f z = {0.f,0.f,0.f,0.f,0.f,0.f,0.f,0.f};
  return z;
}
// async global->LDS copy, 16B per lane, tracked by ASYNCcnt.
// generic LDS pointer low 32 bits == LDS byte address (ISA 10.2 aperture rule).
__device__ __forceinline__ void async_b128(void* lds, const _Float16* g) {
  unsigned l = (unsigned)(unsigned long long)lds;
  unsigned long long ga = (unsigned long long)(const void*)g;
  asm volatile("global_load_async_to_lds_b128 %0, %1, off"
               :: "v"(l), "v"(ga) : "memory");
}

// ---------------------------------------------------------------- kernel 1
// stride-2 subsample + f32->f16 convert: xs[b, i, c] (12544 x 384)
__global__ __launch_bounds__(256)
void k_subsample(const float* __restrict__ x, _Float16* __restrict__ xs) {
  int gid = blockIdx.x * blockDim.x + threadIdx.x;   // over 12544*96
  if (gid >= BATCH * NSUB * (DIM / 4)) return;
  int tok = gid / (DIM / 4);
  int c4  = (gid - tok * (DIM / 4)) * 4;
  int bb  = tok / NSUB;
  int i   = tok - bb * NSUB;
  int iy  = i / 28, ix = i - iy * 28;
  int stok = (iy * 2) * HH + ix * 2;
  const float4 f = *(const float4*)&x[((size_t)bb * NTOK + stok) * DIM + c4];
  v4h h; h.x = (_Float16)f.x; h.y = (_Float16)f.y;
  h.z = (_Float16)f.z; h.w = (_Float16)f.w;
  *(v4h*)&xs[((size_t)bb * NSUB + i) * DIM + c4] = h;
}

// ---------------------------------------------------------------- kernel 1b
__global__ __launch_bounds__(256)
void k_zero16(float4* __restrict__ p, unsigned n16) {
  unsigned gid = blockIdx.x * blockDim.x + threadIdx.x;
  if (gid < n16) { float4 z = {0.f, 0.f, 0.f, 0.f}; p[gid] = z; }
}

// ---------------------------------------------------------------- kernel 1c
// one-time weight f32->f16 convert (row-major [N,K] preserved)
__global__ __launch_bounds__(256)
void k_cvt_w(const float* __restrict__ w, _Float16* __restrict__ wh, int n4) {
  int gid = blockIdx.x * blockDim.x + threadIdx.x;
  if (gid >= n4) return;
  float4 f = ((const float4*)w)[gid];
  v4h h; h.x = (_Float16)f.x; h.y = (_Float16)f.y;
  h.z = (_Float16)f.z; h.w = (_Float16)f.w;
  ((v4h*)wh)[gid] = h;
}

// ---------------------------------------------------------------- GEMM core
// 128x64 macro-tile, K-steps of 32, async-to-LDS double buffering.
// MODE 0: QKV epilogue (scatter q/k rows + v transposed). MODE 1: proj -> f32.
__device__ __forceinline__ void issue_stage(const _Float16* __restrict__ A,
                                            const _Float16* __restrict__ Wh,
                                            _Float16* AsB, _Float16* BsB,
                                            int tm, int tn, int k0, int tid) {
  const int c0 = 2 * tid, c1 = 2 * tid + 1;
  const int ar0 = c0 >> 2, ac0 = (c0 & 3) * 8;
  const int ar1 = c1 >> 2, ac1 = (c1 & 3) * 8;
  const int br  = tid >> 2, bc  = (tid & 3) * 8;
  async_b128(&AsB[ar0 * 40 + ac0], &A[((size_t)(tm + ar0)) * DIM + k0 + ac0]);
  async_b128(&AsB[ar1 * 40 + ac1], &A[((size_t)(tm + ar1)) * DIM + k0 + ac1]);
  async_b128(&BsB[br * 40 + bc],  &Wh[((size_t)(tn + br)) * DIM + k0 + bc]);
}

template <int MODE>
__global__ __launch_bounds__(256)
void k_gemm128(const _Float16* __restrict__ A, const _Float16* __restrict__ Wh,
               const float* __restrict__ bias, _Float16* __restrict__ qh,
               _Float16* __restrict__ kh, _Float16* __restrict__ vt,
               float* __restrict__ out) {
  __shared__ _Float16 As[2][128 * 40];
  __shared__ _Float16 Bs[2][64 * 40];
  const int tid = threadIdx.x, lane = tid & 31, wave = tid >> 5;
  const int hf = lane >> 4, mr = lane & 15;
  const int tm = blockIdx.x * 128, tn = blockIdx.y * 64;
  const int mw = wave >> 1, nw = wave & 1;   // 4 x 2 wave grid of 32x32 tiles

  v8f acc[2][2] = {{zero8(), zero8()}, {zero8(), zero8()}};

  issue_stage(A, Wh, As[0], Bs[0], tm, tn, 0, tid);
  int buf = 0;
  for (int k0 = 0; k0 < DIM; k0 += BK) {
    if (k0 + BK < DIM) {
      issue_stage(A, Wh, As[buf ^ 1], Bs[buf ^ 1], tm, tn, k0 + BK, tid);
      asm volatile("s_wait_asynccnt 0x3" ::: "memory");  // this stage landed
    } else {
      asm volatile("s_wait_asynccnt 0x0" ::: "memory");
    }
    __syncthreads();
    const _Float16* as = As[buf];
    const _Float16* bs = Bs[buf];
    const _Float16* ap0 = &as[(mw * 32 + mr) * 40];
    const _Float16* ap1 = &as[(mw * 32 + 16 + mr) * 40];
    const _Float16* bp0 = &bs[(nw * 32 + mr) * 40 + hf * 16];
    const _Float16* bp1 = &bs[(nw * 32 + 16 + mr) * 40 + hf * 16];
    v16h af0 = frag2(ap0 + hf * 16, ap0 + hf * 16 + 8);
    v16h af1 = frag2(ap1 + hf * 16, ap1 + hf * 16 + 8);
    v16h bf0 = frag2(bp0, bp0 + 8);
    v16h bf1 = frag2(bp1, bp1 + 8);
    acc[0][0] = wmma16(af0, bf0, acc[0][0]);
    acc[0][1] = wmma16(af0, bf1, acc[0][1]);
    acc[1][0] = wmma16(af1, bf0, acc[1][0]);
    acc[1][1] = wmma16(af1, bf1, acc[1][1]);
    __syncthreads();   // protect buf before it is overwritten two stages on
    buf ^= 1;
  }

#pragma unroll
  for (int j = 0; j < 2; ++j) {
    const int gj = tn + nw * 32 + j * 16 + mr;
    const float bv = bias[gj];
    if (MODE == 0) {
      const int part = gj / DIM;
      const int jj = gj - part * DIM;
      const int head = jj / HD;
      const int d = jj - head * HD;
#pragma unroll
      for (int i = 0; i < 2; ++i) {
#pragma unroll
        for (int r = 0; r < 8; ++r) {
          int gm = tm + mw * 32 + i * 16 + r + 8 * hf;
          int bb = gm / NSUB;
          int ii = gm - bb * NSUB;
          size_t bh = (size_t)bb * HEADS + head;
          _Float16 hv = (_Float16)(acc[i][j][r] + bv);
          if (part == 0)      qh[(bh * NPAD + ii) * HDP + d] = hv;
          else if (part == 1) kh[(bh * NPAD + ii) * HDP + d] = hv;
          else                vt[(bh * HDP + d) * NPAD + ii] = hv;
        }
      }
    } else {
#pragma unroll
      for (int i = 0; i < 2; ++i) {
#pragma unroll
        for (int r = 0; r < 8; ++r) {
          int gm = tm + mw * 32 + i * 16 + r + 8 * hf;
          out[(size_t)gm * DIM + gj] = acc[i][j][r] + bv;
        }
      }
    }
  }
}

// ---------------------------------------------------------------- kernel 3
// flash attention: 1 wave per (b, head, 16-query tile); key blocks of 32
__global__ __launch_bounds__(32)
void k_attn(const _Float16* __restrict__ qh, const _Float16* __restrict__ kh,
            const _Float16* __restrict__ vt, float* __restrict__ o) {
  __shared__ _Float16 P[16 * 32];
  const int lane = threadIdx.x & 31;
  const int hf = lane >> 4;
  const int mr = lane & 15;
  int blk = blockIdx.x;
  int qt = blk % QTILES; blk /= QTILES;
  int head = blk % HEADS;
  int bb = blk / HEADS;
  const size_t bh = (size_t)bb * HEADS + head;
  const int qbase = qt * 16;

  // Q fragments 16x64 = two K=32 A-fragments (resident for whole loop)
  const _Float16* qrow = qh + (bh * NPAD + qbase + mr) * HDP;
  v16h qa0 = frag2(qrow + hf * 16,      qrow + 8 + hf * 16);
  v16h qa1 = frag2(qrow + 32 + hf * 16, qrow + 40 + hf * 16);

  v8f oacc[3] = {zero8(), zero8(), zero8()};
  float m[8], l[8];
#pragma unroll
  for (int r = 0; r < 8; ++r) { m[r] = -1e30f; l[r] = 0.f; }

  for (int kb = 0; kb < NPAD; kb += 32) {
    // QK^T: scores 16q x 32keys = two 16x16 tiles, K=64 over d (2 wmma each)
    const _Float16* kp0 = kh + (bh * NPAD + kb + mr) * HDP + hf * 16;
    const _Float16* kp1 = kh + (bh * NPAD + kb + 16 + mr) * HDP + hf * 16;
    __builtin_prefetch(kp0 + 32 * HDP, 0, 1);        // next key block -> L2/L0
    v8f s0 = zero8(), s1 = zero8();
    s0 = wmma16(qa0, frag2(kp0, kp0 + 8), s0);
    s0 = wmma16(qa1, frag2(kp0 + 32, kp0 + 40), s0);
    s1 = wmma16(qa0, frag2(kp1, kp1 + 8), s1);
    s1 = wmma16(qa1, frag2(kp1 + 32, kp1 + 40), s1);

    const bool msk0 = (kb + mr) >= NSUB;
    const bool msk1 = (kb + 16 + mr) >= NSUB;
    float p0a[8], p1a[8];
#pragma unroll
    for (int r = 0; r < 8; ++r) {
      float a = s0[r] * ATT_SCALE; if (msk0) a = -1e30f;
      float c = s1[r] * ATT_SCALE; if (msk1) c = -1e30f;
      float t = fmaxf(a, c);                       // row max over 16-lane half
      t = fmaxf(t, __shfl_xor(t, 1, 16));
      t = fmaxf(t, __shfl_xor(t, 2, 16));
      t = fmaxf(t, __shfl_xor(t, 4, 16));
      t = fmaxf(t, __shfl_xor(t, 8, 16));
      float mn = fmaxf(m[r], t);
      float alpha = __expf(m[r] - mn);
      float e0 = __expf(a - mn), e1 = __expf(c - mn);
      float ps = e0 + e1;                          // row sum
      ps += __shfl_xor(ps, 1, 16);
      ps += __shfl_xor(ps, 2, 16);
      ps += __shfl_xor(ps, 4, 16);
      ps += __shfl_xor(ps, 8, 16);
      l[r] = l[r] * alpha + ps;
      m[r] = mn;
      oacc[0][r] *= alpha; oacc[1][r] *= alpha; oacc[2][r] *= alpha;
      p0a[r] = e0; p1a[r] = e1;
    }
    // restage P (C-layout -> A-layout) through LDS
#pragma unroll
    for (int r = 0; r < 8; ++r) {
      P[(r + 8 * hf) * 32 + mr]      = (_Float16)p0a[r];
      P[(r + 8 * hf) * 32 + 16 + mr] = (_Float16)p1a[r];
    }
    asm volatile("s_wait_dscnt 0" ::: "memory");
    const _Float16* prow = &P[mr * 32];
    v16h pf = frag2(prow + hf * 16, prow + hf * 16 + 8);
    // AV: K=32 (keys), three 16-col tiles over hd
#pragma unroll
    for (int nt = 0; nt < 3; ++nt) {
      const _Float16* vp = vt + (bh * HDP + nt * 16 + mr) * NPAD + kb + hf * 16;
      oacc[nt] = wmma16(pf, frag2(vp, vp + 8), oacc[nt]);
    }
  }

#pragma unroll
  for (int r = 0; r < 8; ++r) {
    float inv = 1.0f / l[r];
    int row = qbase + r + 8 * hf;
    float* op = o + ((size_t)bb * NSUB + row) * DIM + head * HD;
#pragma unroll
    for (int nt = 0; nt < 3; ++nt) op[nt * 16 + mr] = oacc[nt][r] * inv;
  }
}

// ---------------------------------------------------------------- kernel 4
// depthwise 2x2 ConvTranspose upsample + LayerNorm -> y f16 [B, 3136, 384]
__global__ __launch_bounds__(128)
void k_up_ln(const float* __restrict__ o, const float* __restrict__ lp_w,
             const float* __restrict__ lp_b, const float* __restrict__ ln_w,
             const float* __restrict__ ln_b, _Float16* __restrict__ y) {
  __shared__ float red[4][8];
  const int tid = threadIdx.x, lane = tid & 31, wave = tid >> 5;
  const int blk = blockIdx.x;          // b*784 + i
  const int bb = blk / NSUB;
  const int i  = blk - bb * NSUB;
  const int iy = i / 28, ix = i - iy * 28;
  const int c0 = tid * 3;

  float vals[4][3], s1[4] = {0.f,0.f,0.f,0.f}, s2[4] = {0.f,0.f,0.f,0.f};
#pragma unroll
  for (int kk = 0; kk < 3; ++kk) {
    int c = c0 + kk;
    float oc = o[(size_t)blk * DIM + c];
    float lb = lp_b[c];
#pragma unroll
    for (int pq = 0; pq < 4; ++pq) {
      float v = oc * lp_w[c * 4 + pq] + lb;
      vals[pq][kk] = v;
      s1[pq] += v; s2[pq] += v * v;
    }
  }
#pragma unroll
  for (int pq = 0; pq < 4; ++pq) {
#pragma unroll
    for (int off = 16; off > 0; off >>= 1) {
      s1[pq] += __shfl_xor(s1[pq], off);
      s2[pq] += __shfl_xor(s2[pq], off);
    }
  }
  if (lane == 0) {
#pragma unroll
    for (int pq = 0; pq < 4; ++pq) {
      red[wave][pq] = s1[pq]; red[wave][4 + pq] = s2[pq];
    }
  }
  __syncthreads();
#pragma unroll
  for (int pq = 0; pq < 4; ++pq) {
    float t1 = red[0][pq] + red[1][pq] + red[2][pq] + red[3][pq];
    float t2 = red[0][4+pq] + red[1][4+pq] + red[2][4+pq] + red[3][4+pq];
    float mu = t1 * (1.0f / DIM);
    float var = t2 * (1.0f / DIM) - mu * mu;
    float rstd = rsqrtf(var + LN_EPS);
    int p = pq >> 1, q = pq & 1;
    int row = (2 * iy + p) * HH + 2 * ix + q;
    _Float16* yp = y + ((size_t)bb * NTOK + row) * DIM;
#pragma unroll
    for (int kk = 0; kk < 3; ++kk) {
      int c = c0 + kk;
      yp[c] = (_Float16)((vals[pq][kk] - mu) * rstd * ln_w[c] + ln_b[c]);
    }
  }
}

// ---------------------------------------------------------------- launch
extern "C" void kernel_launch(void* const* d_in, const int* in_sizes, int n_in,
                              void* d_out, int out_size, void* d_ws,
                              size_t ws_size, hipStream_t stream) {
  const float* x      = (const float*)d_in[0];
  const float* qkv_w  = (const float*)d_in[1];
  const float* qkv_b  = (const float*)d_in[2];
  const float* proj_w = (const float*)d_in[3];
  const float* proj_b = (const float*)d_in[4];
  const float* lp_w   = (const float*)d_in[5];
  const float* lp_b   = (const float*)d_in[6];
  const float* ln_w   = (const float*)d_in[7];
  const float* ln_b   = (const float*)d_in[8];
  float* out = (float*)d_out;

  char* ws = (char*)d_ws;
  _Float16* xs  = (_Float16*)(ws);                  //  9,633,792 B
  _Float16* qh  = (_Float16*)(ws + 9633792);        // 13,107,200 B
  _Float16* kh  = (_Float16*)(ws + 22740992);       // 13,107,200 B
  _Float16* vt  = (_Float16*)(ws + 35848192);       // 13,107,200 B
  float*    o   = (float*)   (ws + 48955392);       // 19,267,584 B
  _Float16* yh  = (_Float16*)(ws + 68222976);       // 38,535,168 B
  _Float16* whq = (_Float16*)(ws + 106758144);      //    884,736 B
  _Float16* whp = (_Float16*)(ws + 107642880);      //    294,912 B

  // 1) subsample + convert, weight converts, zero q/k/vt pads
  k_subsample<<<(BATCH * NSUB * (DIM / 4) + 255) / 256, 256, 0, stream>>>(x, xs);
  k_cvt_w<<<(110592 + 255) / 256, 256, 0, stream>>>(qkv_w, whq, 110592);
  k_cvt_w<<<(36864 + 255) / 256, 256, 0, stream>>>(proj_w, whp, 36864);
  k_zero16<<<(2457600 + 255) / 256, 256, 0, stream>>>((float4*)qh, 2457600u);
  // 2) QKV GEMM (M=12544 -> 98 tiles of 128, N=1152 -> 18 tiles of 64)
  k_gemm128<0><<<dim3(98, 18), 256, 0, stream>>>(xs, whq, qkv_b, qh, kh, vt,
                                                 nullptr);
  // 3) attention: 16*8*49 waves
  k_attn<<<BATCH * HEADS * QTILES, 32, 0, stream>>>(qh, kh, vt, o);
  // 4) upsample + layernorm
  k_up_ln<<<BATCH * NSUB, 128, 0, stream>>>(o, lp_w, lp_b, ln_w, ln_b, yh);
  // 5) projection GEMM (M=50176 -> 392 tiles of 128, N=384 -> 6 tiles of 64)
  k_gemm128<1><<<dim3(392, 6), 256, 0, stream>>>(yh, whp, proj_b, nullptr,
                                                 nullptr, nullptr, out);
}
